// Cont_InfoNCE_27496380629110
// MI455X (gfx1250) — compile-verified, matrix operand order checked by reading, not run on hardware
//
#include <hip/hip_runtime.h>
#include <hip/hip_bf16.h>

// ---- CDNA5 WMMA types ----
typedef __attribute__((ext_vector_type(16))) __bf16 v16bf;
typedef __attribute__((ext_vector_type(8)))  float  v8f;

#define MDIM 256
#define TDIM 1024
#define T2   2048   // padded length
#define T3   3072   // f2 ring: [f2 | zeros | f2]

// float -> bf16, round-to-nearest-even (inputs are finite)
static __device__ __forceinline__ unsigned short f2bf(float f) {
    unsigned int u = __float_as_uint(f);
    u += 0x7fffu + ((u >> 16) & 1u);
    return (unsigned short)(u >> 16);
}

// ------------------------------------------------------------------
// Kernel 1: center rows, compute std (ddof=1), emit bf16 staging.
//   blocks 0..255   -> zis row b      -> f1c[b][0..T)
//   blocks 256..511 -> zjs row b-256  -> f2d[j] = [f2c, zeros(T), f2c]
// Also zeroes the scalar loss output (stream-ordered before kernel 3).
// ------------------------------------------------------------------
__global__ __launch_bounds__(256)
void prep_kernel(const float* __restrict__ zis, const float* __restrict__ zjs,
                 unsigned short* __restrict__ f1c, unsigned short* __restrict__ f2d,
                 float* __restrict__ s1, float* __restrict__ s2,
                 float* __restrict__ out) {
    __shared__ float red[256];
    const int row  = blockIdx.x;
    const bool isF2 = (row >= MDIM);
    const int r    = isF2 ? row - MDIM : row;
    const float* src = isF2 ? zjs : zis;
    const int tid = threadIdx.x;

    float x[4];
    #pragma unroll
    for (int k = 0; k < 4; ++k) x[k] = src[r * TDIM + tid + 256 * k];

    // mean
    red[tid] = x[0] + x[1] + x[2] + x[3];
    __syncthreads();
    for (int off = 128; off > 0; off >>= 1) {
        if (tid < off) red[tid] += red[tid + off];
        __syncthreads();
    }
    const float mean = red[0] * (1.0f / TDIM);
    __syncthreads();

    float c[4], ssq = 0.f;
    #pragma unroll
    for (int k = 0; k < 4; ++k) { c[k] = x[k] - mean; ssq += c[k] * c[k]; }
    red[tid] = ssq;
    __syncthreads();
    for (int off = 128; off > 0; off >>= 1) {
        if (tid < off) red[tid] += red[tid + off];
        __syncthreads();
    }
    const float sd = sqrtf(red[0] * (1.0f / (TDIM - 1)));

    if (!isF2) {
        #pragma unroll
        for (int k = 0; k < 4; ++k)
            f1c[r * TDIM + tid + 256 * k] = f2bf(c[k]);
        if (tid == 0) s1[r] = sd;
    } else {
        const int base = r * T3;
        #pragma unroll
        for (int k = 0; k < 4; ++k) {
            const int t = tid + 256 * k;
            const unsigned short v = f2bf(c[k]);
            f2d[base + t]        = v;   // u in [0,T)
            f2d[base + TDIM + t] = 0;   // u in [T,2T): zero pad
            f2d[base + T2 + t]   = v;   // u in [2T,3T): wrap copy
        }
        if (tid == 0) s2[r] = sd;
    }
    if (row == 0 && tid == 0) out[0] = 0.0f;
}

// ------------------------------------------------------------------
// Kernel 2: fused pairwise cross-correlation + lag-max via bf16 WMMA.
// Grid (8, 256): block = (32-row i-tile, one j). 4 waves split the 2048
// lags (512 each). 2x2 register blocking: each K-step shares one A pair
// (two 16-row sub-tiles) and one B pair (two lag-tiles) across 4 WMMAs
// -> 2 ds_load_b128 per WMMA instead of 4. Lag-max fused in registers.
// cc[i,j,tau] = sum_t f1[i,t] * f2ring[(t - tau) mod 2T]
//   handled via f2d index u = t - tau + 2T in [1, 3T).
// ------------------------------------------------------------------
__global__ __launch_bounds__(128)
void corr_kernel(const unsigned short* __restrict__ f1c,
                 const unsigned short* __restrict__ f2d,
                 const float* __restrict__ s1, const float* __restrict__ s2,
                 float* __restrict__ dist) {
    __shared__ unsigned short sA[32 * TDIM];   // 64 KB: f1c rows i0..i0+31
    __shared__ unsigned short sB[T3];          // 6 KB : f2 ring for row j
    __shared__ float sRed[4][2][8][32];        // 8 KB : per-wave lane maxima

    const int it = blockIdx.x;
    const int j  = blockIdx.y;
    const int i0 = it * 32;
    const int tid = threadIdx.x;

    // Stage operands into LDS (uint4 = 8 bf16 per transfer)
    {
        const uint4* gA = (const uint4*)(f1c + i0 * TDIM);
        uint4* lA = (uint4*)sA;
        for (int k = tid; k < (32 * TDIM) / 8; k += 128) lA[k] = gA[k];
        const uint4* gB = (const uint4*)(f2d + j * T3);
        uint4* lB = (uint4*)sB;
        for (int k = tid; k < T3 / 8; k += 128) lB[k] = gB[k];
    }
    __syncthreads();

    const int wave = tid >> 5;
    const int lane = tid & 31;
    const int m    = lane & 15;   // A row within 16-row sub-tile
    const int half = lane >> 4;   // K-half selector for 16-bit operands
    const int n    = lane & 15;   // B column = lag within lag-tile

    const unsigned short* aRow0 = sA + m * TDIM;
    const unsigned short* aRow1 = sA + (16 + m) * TDIM;

    float rmax0[8], rmax1[8];
    #pragma unroll
    for (int r = 0; r < 8; ++r) { rmax0[r] = -1e30f; rmax1[r] = -1e30f; }

    for (int lp = 0; lp < 16; ++lp) {
        const int tau0 = wave * 512 + lp * 32 + n;  // lag-tile pair for this wave
        const int ub0  = T2 - tau0;                 // + (kb + k0) -> [1, 3T)
        const int ub1  = ub0 - 16;
        v8f c00 = {0.f,0.f,0.f,0.f,0.f,0.f,0.f,0.f};
        v8f c01 = {0.f,0.f,0.f,0.f,0.f,0.f,0.f,0.f};
        v8f c10 = {0.f,0.f,0.f,0.f,0.f,0.f,0.f,0.f};
        v8f c11 = {0.f,0.f,0.f,0.f,0.f,0.f,0.f,0.f};

        #pragma unroll 2
        for (int kb = 0; kb < TDIM; kb += 32) {
            v16bf a0, a1, b0, b1;
            #pragma unroll
            for (int v = 0; v < 8; ++v) {
                // 16-bit operand K layout: VGPR v holds K = k0, k0+1
                const int k0 = ((v >> 2) << 4) + (half << 3) + ((v & 3) << 1);
                // A: aligned pair loads from LDS (merge to ds_load_b128)
                const unsigned int wa0 = *(const unsigned int*)&aRow0[kb + k0];
                const unsigned int wa1 = *(const unsigned int*)&aRow1[kb + k0];
                a0[2*v]   = __builtin_bit_cast(__bf16, (unsigned short)(wa0 & 0xffffu));
                a0[2*v+1] = __builtin_bit_cast(__bf16, (unsigned short)(wa0 >> 16));
                a1[2*v]   = __builtin_bit_cast(__bf16, (unsigned short)(wa1 & 0xffffu));
                a1[2*v+1] = __builtin_bit_cast(__bf16, (unsigned short)(wa1 >> 16));
                // B: lag-shifted windows of f2 ring (per-lane shift)
                const int u0 = ub0 + kb + k0;
                const int u1 = ub1 + kb + k0;
                b0[2*v]   = __builtin_bit_cast(__bf16, sB[u0]);
                b0[2*v+1] = __builtin_bit_cast(__bf16, sB[u0 + 1]);
                b1[2*v]   = __builtin_bit_cast(__bf16, sB[u1]);
                b1[2*v+1] = __builtin_bit_cast(__bf16, sB[u1 + 1]);
            }
            c00 = __builtin_amdgcn_wmma_f32_16x16x32_bf16(false, a0, false, b0, (short)0, c00, false, false);
            c01 = __builtin_amdgcn_wmma_f32_16x16x32_bf16(false, a0, false, b1, (short)0, c01, false, false);
            c10 = __builtin_amdgcn_wmma_f32_16x16x32_bf16(false, a1, false, b0, (short)0, c10, false, false);
            c11 = __builtin_amdgcn_wmma_f32_16x16x32_bf16(false, a1, false, b1, (short)0, c11, false, false);
        }
        // fuse lag-max into registers: D vgpr r holds row (half*8 + r)
        #pragma unroll
        for (int r = 0; r < 8; ++r) {
            rmax0[r] = fmaxf(rmax0[r], fmaxf(c00[r], c01[r]));
            rmax1[r] = fmaxf(rmax1[r], fmaxf(c10[r], c11[r]));
        }
    }

    #pragma unroll
    for (int r = 0; r < 8; ++r) {
        sRed[wave][0][r][lane] = rmax0[r];
        sRed[wave][1][r][lane] = rmax1[r];
    }
    __syncthreads();

    // Reduce across lanes (16 lag columns) and 4 waves; normalize once.
    if (tid < 32) {
        const int mm  = tid;          // output row within 32-row tile
        const int sub = mm >> 4;      // which 16-row sub-tile
        const int ml  = mm & 15;
        const int r   = ml & 7;       // accumulator vgpr
        const int hh  = ml >> 3;      // lane half holding this row
        float best = -1e30f;
        for (int w = 0; w < 4; ++w)
            #pragma unroll
            for (int l = 0; l < 16; ++l)
                best = fmaxf(best, sRed[w][sub][r][hh * 16 + l]);
        float p = s1[i0 + mm] * s2[j];
        if (p == 0.0f) p = 1.0f;      // reference's power==0 guard
        dist[(i0 + mm) * MDIM + j] = best / (p * (float)(TDIM - 1));
    }
}

// ------------------------------------------------------------------
// Kernel 3: CrossEntropy(reduction='sum') over the 256x256 logits.
// Block per row; loss_i = logsumexp(row) - row[target]; atomicAdd.
// ------------------------------------------------------------------
__global__ __launch_bounds__(256)
void ce_kernel(const float* __restrict__ dist, const int* __restrict__ speeds,
               float* __restrict__ out) {
    __shared__ float red[256];
    const int i = blockIdx.x;
    const int t = threadIdx.x;
    const float v = dist[i * MDIM + t];

    red[t] = v;
    __syncthreads();
    for (int off = 128; off > 0; off >>= 1) {
        if (t < off) red[t] = fmaxf(red[t], red[t + off]);
        __syncthreads();
    }
    const float rmax = red[0];
    __syncthreads();

    red[t] = __expf(v - rmax);
    __syncthreads();
    for (int off = 128; off > 0; off >>= 1) {
        if (t < off) red[t] += red[t + off];
        __syncthreads();
    }
    if (t == 0) {
        const int tgt = speeds[i];
        const float li = logf(red[0]) + rmax - dist[i * MDIM + tgt];
        atomicAdd(out, li);
    }
}

// ------------------------------------------------------------------
extern "C" void kernel_launch(void* const* d_in, const int* in_sizes, int n_in,
                              void* d_out, int out_size, void* d_ws, size_t ws_size,
                              hipStream_t stream) {
    (void)in_sizes; (void)n_in; (void)out_size; (void)ws_size;
    const float* zis    = (const float*)d_in[0];
    const float* zjs    = (const float*)d_in[1];
    const int*   speeds = (const int*)d_in[2];
    float* out = (float*)d_out;

    char* ws = (char*)d_ws;
    unsigned short* f1c = (unsigned short*)(ws);                 // 256*1024*2 = 512 KB
    unsigned short* f2d = (unsigned short*)(ws + 524288);        // 256*3072*2 = 1.5 MB
    float* s1   = (float*)(ws + 2097152);                        // 1 KB
    float* s2   = (float*)(ws + 2098176);                        // 1 KB
    float* dist = (float*)(ws + 2099200);                        // 256*256*4 = 256 KB

    prep_kernel<<<dim3(2 * MDIM), dim3(256), 0, stream>>>(
        zis, zjs, f1c, f2d, s1, s2, out);

    corr_kernel<<<dim3(MDIM / 32, MDIM), dim3(128), 0, stream>>>(
        f1c, f2d, s1, s2, dist);

    ce_kernel<<<dim3(MDIM), dim3(256), 0, stream>>>(dist, speeds, out);
}
